// MixedContrastiveLoss_21560735826020
// MI455X (gfx1250) — matrix operand order repaired and checked by reading, not compiled, not run on hardware
//
#include <hip/hip_runtime.h>

// MI455X / gfx1250 contrastive loss.
// 137 GFLOP GEMM (8192x8192x1024) routed through v_wmma_f32_16x16x32_f16.
// 4x4 register tiles per wave (64x64) -> 16 ds_load_b128 per 16 WMMAs,
// keeping LDS read demand at ~256 B/cycle/WGP (feed-limit), vs 1.5x that
// with 2x4 tiles. Streaming softmax with exact known row-max (self-sim == 1
// -> max = 1/T). No atomics: deterministic per-(colblock,wave) partials.

typedef __attribute__((ext_vector_type(16))) _Float16 v16h;
typedef __attribute__((ext_vector_type(8)))  _Float16 v8h;
typedef __attribute__((ext_vector_type(8)))  float    v8f;

#define NROWS 4096          // N
#define DIM   1024          // D
#define MTOT  8192          // 2N
#define TINV  20.0f         // 1 / temperature

union FragH { v16h v; v8h h[2]; };

// ---------------------------------------------------------------------------
// Kernel 1: normalize rows, emit f16 reps + positive-pair dot products.
// ---------------------------------------------------------------------------
__global__ __launch_bounds__(256) void nrm_kernel(const float* __restrict__ ei,
                                                  const float* __restrict__ ej,
                                                  _Float16* __restrict__ reps,
                                                  float* __restrict__ pos)
{
  const int r = blockIdx.x;
  const int t = threadIdx.x;
  const float* xi = ei + (size_t)r * DIM;
  const float* xj = ej + (size_t)r * DIM;

  float vi[4], vj[4];
  float si = 0.f, sj = 0.f, sd = 0.f;
#pragma unroll
  for (int q = 0; q < 4; ++q) {
    float a = xi[t + q * 256];
    float b = xj[t + q * 256];
    vi[q] = a; vj[q] = b;
    si += a * a; sj += b * b; sd += a * b;
  }
#pragma unroll
  for (int m = 1; m < 32; m <<= 1) {
    si += __shfl_xor(si, m, 32);
    sj += __shfl_xor(sj, m, 32);
    sd += __shfl_xor(sd, m, 32);
  }
  __shared__ float s0[8], s1[8], s2[8];
  const int w = t >> 5;
  if ((t & 31) == 0) { s0[w] = si; s1[w] = sj; s2[w] = sd; }
  __syncthreads();
  if (t == 0) {
    float a = 0.f, b = 0.f, c = 0.f;
#pragma unroll
    for (int i = 0; i < 8; ++i) { a += s0[i]; b += s1[i]; c += s2[i]; }
    float ri = rsqrtf(a), rj = rsqrtf(b);
    s0[0] = ri; s1[0] = rj;
    pos[r] = c * ri * rj;              // cosine of the positive pair
  }
  __syncthreads();
  const float ri = s0[0], rj = s1[0];
#pragma unroll
  for (int q = 0; q < 4; ++q) {
    reps[(size_t)r * DIM + t + q * 256]           = (_Float16)(vi[q] * ri);
    reps[(size_t)(r + NROWS) * DIM + t + q * 256] = (_Float16)(vj[q] * rj);
  }
}

// ---------------------------------------------------------------------------
// Kernel 2: WMMA GEMM tile (128x256 per block, K=1024) + streamed exp-rowsum.
// Grid: (8192/128, 8192/256). 8 waves laid out 2(m) x 4(n); each wave owns a
// 64x64 sub-tile = 4x4 accumulators of v_wmma_f32_16x16x32_f16 (512 WMMA/wave).
// ---------------------------------------------------------------------------
__global__ __launch_bounds__(256) void sim_kernel(const _Float16* __restrict__ reps,
                                                  float* __restrict__ part)
{
  __shared__ _Float16 Als[128][72];   // padded stride: 144B rows, 16B aligned
  __shared__ _Float16 Bls[256][72];

  const int t    = threadIdx.x;
  const int lane = t & 31;
  const int wave = t >> 5;
  const int wm   = wave & 1;          // 64-row group
  const int wn   = wave >> 1;         // 64-col group (0..3)
  const int half = lane >> 4;
  const int l16  = lane & 15;

  const int rowG0 = blockIdx.x * 128;
  const int colG0 = blockIdx.y * 256;

  v8f acc[4][4] = {};

  for (int kb = 0; kb < DIM; kb += 64) {
    // Stage A (128x64) and B (256x64) slabs from reps (sim = reps @ reps^T).
#pragma unroll
    for (int i = 0; i < 4; ++i) {
      int chunk = t + i * 256;              // 0..1023
      int r     = chunk >> 3;               // 0..127
      int off   = (chunk & 7) << 3;         // 0..56 halves (16B aligned)
      *(uint4*)&Als[r][off] =
          *(const uint4*)(reps + (size_t)(rowG0 + r) * DIM + kb + off);
      if (kb + 64 < DIM)
        __builtin_prefetch(reps + (size_t)(rowG0 + r) * DIM + kb + 64 + off, 0, 1);
    }
#pragma unroll
    for (int i = 0; i < 8; ++i) {
      int chunk = t + i * 256;              // 0..2047
      int r     = chunk >> 3;               // 0..255
      int off   = (chunk & 7) << 3;
      *(uint4*)&Bls[r][off] =
          *(const uint4*)(reps + (size_t)(colG0 + r) * DIM + kb + off);
      if (kb + 64 < DIM)
        __builtin_prefetch(reps + (size_t)(colG0 + r) * DIM + kb + 64 + off, 0, 1);
    }
    __syncthreads();

#pragma unroll
    for (int kk = 0; kk < 64; kk += 32) {
      FragH af[4], bf[4];
      // A 16x32 f16 layout: lane = row(l16); VGPR0-3 = K[half*8..+8),
      // VGPR4-7 = K[16+half*8..+8).
#pragma unroll
      for (int tm = 0; tm < 4; ++tm) {
        int row = wm * 64 + tm * 16 + l16;
        af[tm].h[0] = *(const v8h*)&Als[row][kk + half * 8];
        af[tm].h[1] = *(const v8h*)&Als[row][kk + 16 + half * 8];
      }
      // B 32x16 f16 layout: lane = col(l16); VGPR0-3 = K[half*8..+8), VGPR4-7 = K[16+half*8..+8).
#pragma unroll
      for (int tn = 0; tn < 4; ++tn) {
        int col = wn * 64 + tn * 16 + l16;
        bf[tn].h[0] = *(const v8h*)&Bls[col][kk + half * 8];
        bf[tn].h[1] = *(const v8h*)&Bls[col][kk + 16 + half * 8];
      }
#pragma unroll
      for (int tm = 0; tm < 4; ++tm)
#pragma unroll
        for (int tn = 0; tn < 4; ++tn)
          acc[tm][tn] = __builtin_amdgcn_wmma_f32_16x16x32_f16(
              false, af[tm].v, false, bf[tn].v, (short)0, acc[tm][tn],
              false, false);
    }
    __syncthreads();
  }

  // Epilogue: e = exp((sim - 1) * 20)  (row max is exactly 1 -> fixed shift).
  // C layout: VGPR v, lanes 0-15 -> (row v, col l16); lanes 16-31 -> (row v+8).
  float rs[4][8];
#pragma unroll
  for (int tm = 0; tm < 4; ++tm)
#pragma unroll
    for (int v = 0; v < 8; ++v) rs[tm][v] = 0.f;

#pragma unroll
  for (int tm = 0; tm < 4; ++tm)
#pragma unroll
    for (int tn = 0; tn < 4; ++tn)
#pragma unroll
      for (int v = 0; v < 8; ++v)
        rs[tm][v] += __expf((acc[tm][tn][v] - 1.0f) * TINV);

  const int slab = blockIdx.y * 4 + wn;     // 32*4 = 128 deterministic slabs
#pragma unroll
  for (int tm = 0; tm < 4; ++tm)
#pragma unroll
    for (int v = 0; v < 8; ++v) {
      float s = rs[tm][v];
      s += __shfl_xor(s, 1, 16);
      s += __shfl_xor(s, 2, 16);
      s += __shfl_xor(s, 4, 16);
      s += __shfl_xor(s, 8, 16);
      if (l16 == 0) {
        int row = rowG0 + wm * 64 + tm * 16 + half * 8 + v;
        part[(size_t)slab * MTOT + row] = s;   // unique writer per slot
      }
    }
}

// ---------------------------------------------------------------------------
// Kernel 3: combine partials -> loss = 1/T + mean(log S) - sum(pos)/(N*T).
// ---------------------------------------------------------------------------
__global__ __launch_bounds__(256) void fin_kernel(const float* __restrict__ part,
                                                  const float* __restrict__ pos,
                                                  float* __restrict__ out)
{
  const int t = threadIdx.x;
  float accL = 0.f, accP = 0.f;
  for (int r = t; r < MTOT; r += 256) {
    float s = 0.f;
    for (int p = 0; p < 128; ++p) s += part[(size_t)p * MTOT + r];
    accL += logf(s);
  }
  for (int r = t; r < NROWS; r += 256) accP += pos[r];
#pragma unroll
  for (int m = 1; m < 32; m <<= 1) {
    accL += __shfl_xor(accL, m, 32);
    accP += __shfl_xor(accP, m, 32);
  }
  __shared__ float sL[8], sP[8];
  if ((t & 31) == 0) { sL[t >> 5] = accL; sP[t >> 5] = accP; }
  __syncthreads();
  if (t == 0) {
    float L = 0.f, P = 0.f;
#pragma unroll
    for (int i = 0; i < 8; ++i) { L += sL[i]; P += sP[i]; }
    out[0] = TINV + L / (float)MTOT - P * TINV / (float)NROWS;
  }
}

// ---------------------------------------------------------------------------
extern "C" void kernel_launch(void* const* d_in, const int* in_sizes, int n_in,
                              void* d_out, int out_size, void* d_ws, size_t ws_size,
                              hipStream_t stream) {
  (void)in_sizes; (void)n_in; (void)out_size; (void)ws_size;
  const float* emb_i = (const float*)d_in[0];
  const float* emb_j = (const float*)d_in[1];

  // Workspace layout (all fully written before read; no init needed):
  //   [0, 16MB)   : f16 reps (8192 x 1024)
  //   [16, 20MB)  : f32 partial rowsums (128 x 8192)
  //   [20MB, +16K): f32 pos dots (4096)
  _Float16* reps = (_Float16*)d_ws;
  float* part = (float*)((char*)d_ws + (size_t)MTOT * DIM * sizeof(_Float16));
  float* pos  = (float*)((char*)part + (size_t)128 * MTOT * sizeof(float));

  nrm_kernel<<<NROWS, 256, 0, stream>>>(emb_i, emb_j, reps, pos);
  dim3 grid(MTOT / 128, MTOT / 256);
  sim_kernel<<<grid, 256, 0, stream>>>(reps, part);
  fin_kernel<<<1, 256, 0, stream>>>(part, pos, (float*)d_out);
}
